// MaskPredictor_86938728006271
// MI455X (gfx1250) — matrix operand order repaired and checked by reading, not compiled, not run on hardware
//
#include <hip/hip_runtime.h>
#include <math.h>

typedef __attribute__((ext_vector_type(16))) __bf16 v16bf;
typedef __attribute__((ext_vector_type(8)))  float  v8f;

union V16 { v16bf v; __bf16 e[16]; unsigned short u[16]; };

// Exact gelu (erf form) — used only in the small prep kernels.
__device__ __forceinline__ float gelu_f(float x) {
    return 0.5f * x * (1.0f + erff(x * 0.70710678118654752f));
}

// Branchless fast gelu (tanh form) for the hot kernel: native v_exp_f32 + v_rcp_f32,
// no EXEC divergence between WMMA bursts.
__device__ __forceinline__ float gelu_fast(float x) {
    float u = 0.7978845608f * fmaf(0.044715f * x, x * x, x);   // sqrt(2/pi)*(x+0.044715x^3)
    float a = fminf(fmaxf(2.0f * u, -40.0f), 40.0f);
    float t = __expf(a);                                       // v_exp_f32 path
    float r = __builtin_amdgcn_rcpf(t + 1.0f);                 // v_rcp_f32
    float th = fmaf(-2.0f, r, 1.0f);                           // tanh(u)
    return 0.5f * x * (1.0f + th);
}

// ---------------- K0: layernorm + x@W_in + gelu -> h ----------------
__global__ __launch_bounds__(256) void k_ln_gemm(
    const float* __restrict__ x, const float* __restrict__ ln_g,
    const float* __restrict__ ln_b, const float* __restrict__ W_in,
    const float* __restrict__ b_in, float* __restrict__ hbuf)
{
    __shared__ float red[256];
    __shared__ float lnr[256];
    int tid = threadIdx.x;
    size_t row = blockIdx.x;                    // b*1024 + j, 0..2047
    float v = x[row * 256 + tid];

    red[tid] = v; __syncthreads();
    for (int s = 128; s > 0; s >>= 1) { if (tid < s) red[tid] += red[tid + s]; __syncthreads(); }
    float mu = red[0] * (1.0f / 256.0f);
    __syncthreads();
    float d = v - mu;
    red[tid] = d * d; __syncthreads();
    for (int s = 128; s > 0; s >>= 1) { if (tid < s) red[tid] += red[tid + s]; __syncthreads(); }
    float var = red[0] * (1.0f / 256.0f);
    __syncthreads();
    lnr[tid] = d * rsqrtf(var + 1e-5f) * ln_g[tid] + ln_b[tid];
    __syncthreads();

    float acc = 0.0f;
    for (int k = 0; k < 256; ++k)
        acc += lnr[k] * W_in[k * 256 + tid];    // coalesced over tid
    hbuf[row * 256 + tid] = gelu_f(acc + b_in[tid]);
}

// ---------------- K1: glob = sum_j h[:,:,128:]*policy / sum(policy) ----------------
__global__ __launch_bounds__(128) void k_glob(
    const float* __restrict__ hbuf, const float* __restrict__ policy,
    float* __restrict__ glob)
{
    int b = blockIdx.x, hc = threadIdx.x;       // 128 threads
    float acc = 0.0f, ps = 0.0f;
    for (int j = 0; j < 1024; ++j) {
        float p = policy[b * 1024 + j];
        acc += hbuf[((size_t)b * 1024 + j) * 256 + 128 + hc] * p;
        ps += p;
    }
    glob[b * 128 + hc] = acc / ps;
}

// ---------------- K2: A[b,j] = h[b,j,:128] @ W1[0:128,:] ----------------
__global__ __launch_bounds__(256) void k_local_w1(
    const float* __restrict__ hbuf, const float* __restrict__ W1,
    float* __restrict__ Arows)
{
    __shared__ float hl[128];
    int tid = threadIdx.x;
    size_t row = blockIdx.x;
    if (tid < 128) hl[tid] = hbuf[row * 256 + tid];
    __syncthreads();
    float acc = 0.0f;
    for (int k = 0; k < 128; ++k)
        acc += hl[k] * W1[k * 256 + tid];
    Arows[row * 256 + tid] = acc;
}

// ---------------- K3: qb[b,i] = q@W1[256:512] + glob@W1[128:256] + b1 ----------------
__global__ __launch_bounds__(256) void k_qb(
    const float* __restrict__ query, const float* __restrict__ glob,
    const float* __restrict__ W1, const float* __restrict__ b1,
    float* __restrict__ qb)
{
    __shared__ float qr[256];
    __shared__ float gl[128];
    int tid = threadIdx.x;
    int by = blockIdx.x;                        // b*100 + i
    int b = by / 100, i = by % 100;
    qr[tid] = query[((size_t)i * 2 + b) * 256 + tid];   // query is (n,B,C)
    if (tid < 128) gl[tid] = glob[b * 128 + tid];
    __syncthreads();
    float acc = b1[tid];
    for (int k = 0; k < 128; ++k) acc += gl[k] * W1[(128 + k) * 256 + tid];
    for (int k = 0; k < 256; ++k) acc += qr[k] * W1[(256 + k) * 256 + tid];
    qb[(size_t)by * 256 + tid] = acc;
}

// ---------------- K4: pack W2 (256x128 f32) into bf16 WMMA B-fragment order ----------------
// B of v_wmma_f32_16x16x32_bf16 (32x16): lane = col%16 + 16*((kin/8)%2), elem h = kin%8 + 8*(kin/16).
__global__ __launch_bounds__(256) void k_w2pack(
    const float* __restrict__ W2, unsigned short* __restrict__ w2f)
{
    int idx = blockIdx.x * 256 + threadIdx.x;   // 32768 elements
    int k = idx >> 7, nn = idx & 127;
    int kc = k >> 5, kin = k & 31;
    int t = nn >> 4, l16 = nn & 15;
    int lane = l16 + (((kin >> 3) & 1) << 4);
    int h = (kin & 7) + ((kin >> 4) << 3);
    __bf16 bv = (__bf16)W2[idx];
    w2f[((kc * 8 + t) * 32 + lane) * 16 + h] = *(unsigned short*)&bv;
}

// ---------------- K5: fused main — gelu(A+qb) @ W2 (WMMA bf16) -> gelu -> @W3 -> log_softmax ----------------
__global__ __launch_bounds__(256) void k_main(
    const float* __restrict__ Arows, const float* __restrict__ qb,
    const unsigned short* __restrict__ w2f,
    const float* __restrict__ b2, const float* __restrict__ W3,
    const float* __restrict__ b3, float* __restrict__ out)
{
    __shared__ float qbs[256];
    __shared__ float w3s[256];
    __shared__ float b2s[128];
    __shared__ float b3s[2];

    int tid = threadIdx.x;
    int by = blockIdx.y;                        // b*100 + i
    int b = by / 100;
    qbs[tid] = qb[(size_t)by * 256 + tid];
    w3s[tid] = W3[tid];                         // W3 is (128,2)
    if (tid < 128) b2s[tid] = b2[tid];
    if (tid < 2)   b3s[tid] = b3[tid];
    __syncthreads();

    int wave   = tid >> 5;
    int lane   = tid & 31;
    int l16    = lane & 15;
    int laneHi = lane >> 4;
    int j0  = (blockIdx.x * 8 + wave) * 16;     // 16-row tile start in N
    int row = j0 + l16;
    const float* arow = Arows + ((size_t)b * 1024 + row) * 256;

    // Build 8 bf16 A-fragments (ISA 16-bit A 16x32 layout: lo lanes K in {0..7,16..23}+...)
    V16 af[8];
    #pragma unroll
    for (int kc = 0; kc < 8; ++kc) {
        int kb = kc * 32 + laneHi * 8;
        float4 a0 = *(const float4*)(arow + kb);
        float4 a1 = *(const float4*)(arow + kb + 4);
        float4 a2 = *(const float4*)(arow + kb + 16);
        float4 a3 = *(const float4*)(arow + kb + 20);
        float v[16] = {a0.x,a0.y,a0.z,a0.w, a1.x,a1.y,a1.z,a1.w,
                       a2.x,a2.y,a2.z,a2.w, a3.x,a3.y,a3.z,a3.w};
        #pragma unroll
        for (int h = 0; h < 16; ++h) {
            int k = kb + (h & 7) + 16 * (h >> 3);
            af[kc].e[h] = (__bf16)gelu_fast(v[h] + qbs[k]);   // branchless, native cvt
        }
    }

    // GEMM2 via WMMA; epilogue folds gelu + (128->2) GEMM3 into per-lane partials.
    // t-loop deliberately NOT unrolled: keeps live VGPRs < 256 (no s_set_vgpr_msb,
    // better occupancy to hide the L2 B-fragment loads and the WMMA accumulate chain).
    float p0[8] = {}, p1[8] = {};
    #pragma unroll 1
    for (int t = 0; t < 8; ++t) {
        v8f acc = {0.f,0.f,0.f,0.f,0.f,0.f,0.f,0.f};
        #pragma unroll
        for (int kc = 0; kc < 8; ++kc) {
            V16 bf;
            bf.v = *(const v16bf*)(w2f + ((kc * 8 + t) * 32 + lane) * 16);
            acc = __builtin_amdgcn_wmma_f32_16x16x32_bf16(
                      false, af[kc].v, false, bf.v, (short)0, acc, false, false);
        }
        int col = t * 16 + l16;
        float w30 = w3s[col * 2], w31 = w3s[col * 2 + 1], bb = b2s[col];
        #pragma unroll
        for (int r = 0; r < 8; ++r) {
            float z = gelu_fast(acc[r] + bb);   // C/D layout: VGPR r = row r (+8 in hi lanes)
            p0[r] = fmaf(z, w30, p0[r]);
            p1[r] = fmaf(z, w31, p1[r]);
        }
    }

    // Reduce across the 16 lanes of each half (cols), keep halves separate (xor < 16).
    #pragma unroll
    for (int r = 0; r < 8; ++r) {
        #pragma unroll
        for (int m = 1; m < 16; m <<= 1) {
            p0[r] += __shfl_xor(p0[r], m, 32);
            p1[r] += __shfl_xor(p1[r], m, 32);
        }
    }

    if (l16 == 0) {
        float bb0 = b3s[0], bb1 = b3s[1];
        #pragma unroll
        for (int r = 0; r < 8; ++r) {
            int rr = j0 + laneHi * 8 + r;
            float z0 = p0[r] + bb0, z1 = p1[r] + bb1;
            float mx = fmaxf(z0, z1);
            float d  = fminf(z0, z1) - mx;                    // <= 0
            float ls = mx + __logf(1.0f + __expf(d));         // branchless log-sum-exp
            size_t o = ((size_t)by * 1024 + rr) * 2;
            out[o]     = z0 - ls;
            out[o + 1] = z1 - ls;
        }
    }
}

extern "C" void kernel_launch(void* const* d_in, const int* in_sizes, int n_in,
                              void* d_out, int out_size, void* d_ws, size_t ws_size,
                              hipStream_t stream) {
    const float* x      = (const float*)d_in[0];
    const float* query  = (const float*)d_in[1];
    const float* policy = (const float*)d_in[2];
    const float* ln_g   = (const float*)d_in[3];
    const float* ln_b   = (const float*)d_in[4];
    const float* W_in   = (const float*)d_in[5];
    const float* b_in   = (const float*)d_in[6];
    const float* W1     = (const float*)d_in[7];
    const float* b1     = (const float*)d_in[8];
    const float* W2     = (const float*)d_in[9];
    const float* b2     = (const float*)d_in[10];
    const float* W3     = (const float*)d_in[11];
    const float* b3     = (const float*)d_in[12];
    float* out = (float*)d_out;

    // workspace layout (floats): h[2048*256] | glob[256] | A[2048*256] | qb[200*256] | w2f(bf16)[32768]
    float* hbuf  = (float*)d_ws;
    float* glob  = hbuf + 2048 * 256;
    float* Arows = glob + 256;
    float* qbuf  = Arows + 2048 * 256;
    unsigned short* w2f = (unsigned short*)(qbuf + 200 * 256);  // 32B-aligned

    k_ln_gemm<<<2048, 256, 0, stream>>>(x, ln_g, ln_b, W_in, b_in, hbuf);
    k_glob<<<2, 128, 0, stream>>>(hbuf, policy, glob);
    k_local_w1<<<2048, 256, 0, stream>>>(hbuf, W1, Arows);
    k_qb<<<200, 256, 0, stream>>>(query, glob, W1, b1, qbuf);
    k_w2pack<<<128, 256, 0, stream>>>(W2, w2f);
    k_main<<<dim3(8, 200), 256, 0, stream>>>(Arows, qbuf, w2f, b2, W3, b3, out);
}
